// MSA_35175782154895
// MI455X (gfx1250) — matrix operand order repaired and checked by reading, not compiled, not run on hardware
//
#include <hip/hip_runtime.h>

// ---------- types ----------
typedef __bf16 bf16_t;
typedef __attribute__((ext_vector_type(16))) __bf16 v16bf;
typedef __attribute__((ext_vector_type(8)))  float  v8f;
typedef __attribute__((ext_vector_type(4)))  int    v4i;

#define D_MODEL 768
#define SEQ     2048
#define NBATCH  2
#define NTOK    (NBATCH * SEQ)     // 4096
#define NHEAD   12
#define DK      64
#define BH      (NBATCH * NHEAD)   // 24

// ---------- optional CDNA5 async Global->LDS path (probe via __has_builtin) ----
#if __has_builtin(__builtin_amdgcn_global_load_async_to_lds_b128) && \
    __has_builtin(__builtin_amdgcn_s_wait_asynccnt)
#define HAVE_ASYNC_LDS 1
#endif

typedef __attribute__((address_space(1))) v4i* gptr_v4i;   // global int4*
typedef __attribute__((address_space(3))) v4i* lptr_v4i;   // LDS int4*

__device__ __forceinline__ void copy16_g2l(const void* g, void* l) {
#ifdef HAVE_ASYNC_LDS
    __builtin_amdgcn_global_load_async_to_lds_b128((gptr_v4i)g, (lptr_v4i)l, 0, 0);
#else
    *(uint4*)l = *(const uint4*)g;           // fallback: through VGPRs
#endif
}
__device__ __forceinline__ void wait_async_lds() {
#ifdef HAVE_ASYNC_LDS
    __builtin_amdgcn_s_wait_asynccnt(0);
#else
    __syncthreads();
#endif
}
__device__ __forceinline__ void wait_lds_reads() {
#if __has_builtin(__builtin_amdgcn_s_wait_dscnt)
    __builtin_amdgcn_s_wait_dscnt(0);
#else
    __syncthreads();
#endif
}

// ---------- WMMA fragment index maps (wave32; A/B use this code's contiguous k order) ----------
// A (16x32 bf16): lane -> M = lane&15 ; element i -> K = i + hi*16
__device__ __forceinline__ int a_k(int i, int hi) {
    return i + hi * 16;
}
// B (32x16 bf16): lane -> N = lane&15 ; element i -> K = i + hi*16
__device__ __forceinline__ int b_k(int i, int hi) { return i + hi * 16; }
// C/D (16x16 f32): lane -> N = lane&15 ; vgpr r -> M = r + hi*8

// =====================================================================
// Kernel 1: qkv = x @ w_qkv.T, scattered into Q/K/V [bh, n, dk] (bf16)
// column decode: j = qkv*768 + dk*12 + head   (reference's reshape order)
// =====================================================================
__global__ void __launch_bounds__(128)
qkv_gemm_kernel(const float* __restrict__ x, const float* __restrict__ wqkv,
                bf16_t* __restrict__ Qb, bf16_t* __restrict__ Kb, bf16_t* __restrict__ Vb) {
    const int NT_N = (3 * D_MODEL) / 16;                 // 144 col tiles
    const int wave = blockIdx.x * (blockDim.x >> 5) + (threadIdx.x >> 5);
    if (wave >= (NTOK / 16) * NT_N) return;
    const int mt = wave / NT_N;
    const int nt = wave % NT_N;
    const int lane = threadIdx.x & 31;
    const int l16  = lane & 15;
    const int hi   = lane >> 4;

    const float* xrow = x + (size_t)(mt * 16 + l16) * D_MODEL;       // A row (M = l16)
    const float* wrow = wqkv + (size_t)(nt * 16 + l16) * D_MODEL;    // B col (N = l16)

    v8f c = {};
    for (int kk = 0; kk < D_MODEL; kk += 32) {
        __builtin_prefetch(xrow + kk + 32, 0, 0);
        __builtin_prefetch(wrow + kk + 32, 0, 0);
        v16bf a, b;
#pragma unroll
        for (int i = 0; i < 16; i += 2) {                // A: contiguous pairs
            float2 t = *(const float2*)(xrow + kk + a_k(i, hi));
            a[i] = (bf16_t)t.x; a[i + 1] = (bf16_t)t.y;
        }
        const float* wp = wrow + kk + hi * 16;           // B: 16 contiguous floats
#pragma unroll
        for (int i = 0; i < 16; i += 4) {
            float4 t = *(const float4*)(wp + i);
            b[i] = (bf16_t)t.x; b[i+1] = (bf16_t)t.y; b[i+2] = (bf16_t)t.z; b[i+3] = (bf16_t)t.w;
        }
        c = __builtin_amdgcn_wmma_f32_16x16x32_bf16(false, a, false, b, (short)0, c, false, false);
    }

    // scatter through the (qkv, dk, head) permutation
    const int j    = nt * 16 + l16;        // global output column of this lane
    const int qsel = j / D_MODEL;
    const int rem  = j % D_MODEL;
    const int dk   = rem / NHEAD;
    const int hh   = rem % NHEAD;
    bf16_t* dst = (qsel == 0) ? Qb : (qsel == 1) ? Kb : Vb;
#pragma unroll
    for (int r = 0; r < 8; ++r) {
        const int tok = mt * 16 + r + hi * 8;            // 0..4095
        const int bb = tok / SEQ, nn = tok % SEQ;
        dst[((size_t)(bb * NHEAD + hh) * SEQ + nn) * DK + dk] = (bf16_t)c[r];
    }
}

// =====================================================================
// Kernel 2: flash attention, one wave per (bh, 16-row block)
// S = (Q K^T) * 8 ; online softmax ; O += P V ; O -> [b, n, h*dk] bf16
// V tiles streamed Global->LDS via async-copy (ASYNCcnt), overlapped
// with the QK^T WMMAs and softmax.
// =====================================================================
__global__ void __launch_bounds__(32)
attn_kernel(const bf16_t* __restrict__ Qb, const bf16_t* __restrict__ Kb,
            const bf16_t* __restrict__ Vb, bf16_t* __restrict__ Ob) {
    __shared__ bf16_t p_lds[16][36];                     // P tile staging (C->A relayout)
    __shared__ bf16_t v_lds[32][DK];                     // V tile: 32 tokens x 64, 4 KB

    const int bh = blockIdx.x >> 7;                      // / (SEQ/16)
    const int mt = blockIdx.x & 127;
    const int lane = threadIdx.x & 31;
    const int l16  = lane & 15;
    const int hi   = lane >> 4;

    const bf16_t* Qh = Qb + (size_t)bh * SEQ * DK;
    const bf16_t* Kh = Kb + (size_t)bh * SEQ * DK;
    const bf16_t* Vh = Vb + (size_t)bh * SEQ * DK;

    // Q tile 16x64 as two A fragments (reused for all column blocks)
    v16bf qa[2];
    {
        const bf16_t* qrow = Qh + (size_t)(mt * 16 + l16) * DK;
#pragma unroll
        for (int ks = 0; ks < 2; ++ks)
#pragma unroll
            for (int i = 0; i < 16; ++i)
                qa[ks][i] = qrow[ks * 32 + a_k(i, hi)];
    }

    float mrow[8], lrow[8];
    v8f acc[4] = {};
#pragma unroll
    for (int r = 0; r < 8; ++r) { mrow[r] = -1e30f; lrow[r] = 0.0f; }

    for (int j0 = 0; j0 < SEQ; j0 += 32) {
        // ---- kick off async DMA of the V tile (32 tokens x 128 B) into LDS.
        // Make sure the previous iteration's LDS reads have drained first.
        wait_lds_reads();
#pragma unroll
        for (int it = 0; it < 8; ++it) {                 // 8 x (32 lanes x 16 B) = 4 KB
            const int chunk = it * 32 + lane;            // 0..255
            const int row   = chunk >> 3;                // token within tile
            const int c16   = (chunk & 7) * 8;           // bf16 column of 16-byte piece
            copy16_g2l(Vh + (size_t)(j0 + row) * DK + c16, &v_lds[row][c16]);
        }

        // ---- S = Q K^T over two 16-col tiles, k = 64 in two steps of 32
        v8f s0 = {}, s1 = {};
#pragma unroll
        for (int ks = 0; ks < 2; ++ks) {
            v16bf b0, b1;
            const bf16_t* k0 = Kh + (size_t)(j0 + l16) * DK + ks * 32 + hi * 16;
            const bf16_t* k1 = Kh + (size_t)(j0 + 16 + l16) * DK + ks * 32 + hi * 16;
#pragma unroll
            for (int i = 0; i < 16; ++i) { b0[i] = k0[i]; b1[i] = k1[i]; }
            s0 = __builtin_amdgcn_wmma_f32_16x16x32_bf16(false, qa[ks], false, b0, (short)0, s0, false, false);
            s1 = __builtin_amdgcn_wmma_f32_16x16x32_bf16(false, qa[ks], false, b1, (short)0, s1, false, false);
        }

        // ---- online softmax (temperature MULTIPLIES: *8 = sqrt(64))
        float p0[8], p1[8], scale[8];
#pragma unroll
        for (int r = 0; r < 8; ++r) {
            float t0 = s0[r] * 8.0f, t1 = s1[r] * 8.0f;
            float mx = fmaxf(t0, t1);
#pragma unroll
            for (int off = 1; off < 16; off <<= 1)       // stays inside 16-lane half
                mx = fmaxf(mx, __shfl_xor(mx, off, 32));
            const float mn = fmaxf(mrow[r], mx);
            scale[r] = __expf(mrow[r] - mn);
            mrow[r]  = mn;
            p0[r] = __expf(t0 - mn);
            p1[r] = __expf(t1 - mn);
            float rs = p0[r] + p1[r];
#pragma unroll
            for (int off = 1; off < 16; off <<= 1)
                rs += __shfl_xor(rs, off, 32);
            lrow[r] = lrow[r] * scale[r] + rs;
        }
#pragma unroll
        for (int ch = 0; ch < 4; ++ch)
#pragma unroll
            for (int r = 0; r < 8; ++r) acc[ch][r] *= scale[r];

        // ---- P: C-layout -> LDS -> A-layout (bf16)
        __syncthreads();                                  // single-wave WG: lowers to NOP+wait
#pragma unroll
        for (int r = 0; r < 8; ++r) {
            p_lds[r + hi * 8][l16]      = (bf16_t)p0[r];
            p_lds[r + hi * 8][16 + l16] = (bf16_t)p1[r];
        }
        __syncthreads();
        v16bf pa;
#pragma unroll
        for (int i = 0; i < 16; ++i) pa[i] = p_lds[l16][a_k(i, hi)];

        // ---- wait for the V tile DMA, then O += P (16x32) @ V (32x64)
        wait_async_lds();
#pragma unroll
        for (int ch = 0; ch < 4; ++ch) {
            v16bf vb;
#pragma unroll
            for (int i = 0; i < 16; ++i)
                vb[i] = v_lds[i + hi * 16][ch * 16 + l16];
            acc[ch] = __builtin_amdgcn_wmma_f32_16x16x32_bf16(false, pa, false, vb, (short)0, acc[ch], false, false);
        }
    }

    // ---- epilogue: normalize and store to [b, n, head*64 + d]
    const int bb = bh / NHEAD, hh = bh % NHEAD;
#pragma unroll
    for (int ch = 0; ch < 4; ++ch)
#pragma unroll
        for (int r = 0; r < 8; ++r) {
            const int tok = mt * 16 + r + hi * 8;
            const float v = acc[ch][r] / lrow[r];
            Ob[(size_t)(bb * SEQ + tok) * D_MODEL + hh * DK + ch * 16 + l16] = (bf16_t)v;
        }
}

// =====================================================================
// Kernel 3: out = attn_out @ w_fc.T + b_fc  (f32 output)
// =====================================================================
__global__ void __launch_bounds__(128)
fc_gemm_kernel(const bf16_t* __restrict__ A, const float* __restrict__ wfc,
               const float* __restrict__ bias, float* __restrict__ out) {
    const int NT_N = D_MODEL / 16;                       // 48
    const int wave = blockIdx.x * (blockDim.x >> 5) + (threadIdx.x >> 5);
    if (wave >= (NTOK / 16) * NT_N) return;
    const int mt = wave / NT_N, nt = wave % NT_N;
    const int lane = threadIdx.x & 31;
    const int l16  = lane & 15;
    const int hi   = lane >> 4;

    const bf16_t* arow = A + (size_t)(mt * 16 + l16) * D_MODEL;
    const float*  wrow = wfc + (size_t)(nt * 16 + l16) * D_MODEL;

    v8f c = {};
    for (int kk = 0; kk < D_MODEL; kk += 32) {
        __builtin_prefetch(arow + kk + 32, 0, 0);
        __builtin_prefetch(wrow + kk + 32, 0, 0);
        v16bf a, b;
#pragma unroll
        for (int i = 0; i < 16; ++i) a[i] = arow[kk + a_k(i, hi)];
        const float* wp = wrow + kk + hi * 16;
#pragma unroll
        for (int i = 0; i < 16; i += 4) {
            float4 t = *(const float4*)(wp + i);
            b[i] = (bf16_t)t.x; b[i+1] = (bf16_t)t.y; b[i+2] = (bf16_t)t.z; b[i+3] = (bf16_t)t.w;
        }
        c = __builtin_amdgcn_wmma_f32_16x16x32_bf16(false, a, false, b, (short)0, c, false, false);
    }

    const int ncol = nt * 16 + l16;
    const float bc = bias[ncol];
#pragma unroll
    for (int r = 0; r < 8; ++r) {
        const int tok = mt * 16 + r + hi * 8;
        out[(size_t)tok * D_MODEL + ncol] = c[r] + bc;
    }
}

// =====================================================================
extern "C" void kernel_launch(void* const* d_in, const int* in_sizes, int n_in,
                              void* d_out, int out_size, void* d_ws, size_t ws_size,
                              hipStream_t stream) {
    const float* x    = (const float*)d_in[0];   // [2, 2048, 768]
    const float* wqkv = (const float*)d_in[1];   // [2304, 768]
    const float* wfc  = (const float*)d_in[2];   // [768, 768]
    const float* bfc  = (const float*)d_in[3];   // [768]
    // d_in[4] = n_head scalar (12, baked into the kernels)

    // workspace: Q, K, V [bh, n, dk] bf16 + attn-out [b, n, d] bf16  (~25.2 MB)
    const size_t qkv_elems = (size_t)BH * SEQ * DK;      // 3,145,728 each
    bf16_t* Qb = (bf16_t*)d_ws;
    bf16_t* Kb = Qb + qkv_elems;
    bf16_t* Vb = Kb + qkv_elems;
    bf16_t* Ob = Vb + qkv_elems;                         // [2, 2048, 768]

    {   // 256 * 144 = 36864 tiles, 4 waves / block
        const int tiles = (NTOK / 16) * ((3 * D_MODEL) / 16);
        qkv_gemm_kernel<<<tiles / 4, 128, 0, stream>>>(x, wqkv, Qb, Kb, Vb);
    }
    {   // 24 bh * 128 row-blocks, one wave each
        attn_kernel<<<BH * (SEQ / 16), 32, 0, stream>>>(Qb, Kb, Vb, Ob);
    }
    {   // 256 * 48 = 12288 tiles, 4 waves / block
        const int tiles = (NTOK / 16) * (D_MODEL / 16);
        fc_gemm_kernel<<<tiles / 4, 128, 0, stream>>>(Ob, wfc, bfc, (float*)d_out);
    }
}
